// MultiHeadAttention_19000935317791
// MI455X (gfx1250) — matrix operand order repaired and checked by reading, not compiled, not run on hardware
//
#include <hip/hip_runtime.h>

#define D_MODEL 1024
#define T_SEQ   2048
#define NH      16
#define DK      64

typedef __attribute__((ext_vector_type(16))) _Float16 v16h;
typedef __attribute__((ext_vector_type(8)))  _Float16 v8h;
typedef __attribute__((ext_vector_type(4)))  _Float16 v4h;
typedef __attribute__((ext_vector_type(8)))  float    v8f;

union V16U { v16h v; v8h h[2]; };

__device__ __forceinline__ v8f wmma16(v16h a, v16h b, v8f c) {
    // (neg_a, A, neg_b, B, c_mod, C, reuse_a, reuse_b)
    return __builtin_amdgcn_wmma_f32_16x16x32_f16(false, a, false, b, (short)0, c,
                                                  false, false);
}

// A-fragment (16x32 f16, M x K), row-major source, leading dim `ld` (f16 units).
// Lane: M = lane&15; lanes 0-15 hold K=0..7 & 16..23, lanes 16-31 hold K=8..15 & 24..31.
__device__ __forceinline__ v16h frag_a(const _Float16* base, int ld) {
    const int lane = threadIdx.x & 31;
    const int m  = lane & 15;
    const int kb = (lane >> 4) << 3;           // 0 or 8
    const _Float16* p = base + m * ld + kb;
    V16U r;
    r.h[0] = *(const v8h*)p;                   // K = kb .. kb+7
    r.h[1] = *(const v8h*)(p + 16);            // K = kb+16 .. kb+23
    return r.v;
}

// B-fragment (32x16 f16, K x N) from an N-major source: lanes 0-15 hold
// K=0..7 & 16..23, lanes 16-31 hold K=8..15 & 24..31 (src[n*ld + k]).
__device__ __forceinline__ v16h frag_b(const _Float16* base, int ld) {
    const int lane = threadIdx.x & 31;
    const int n  = lane & 15;
    const int kb = (lane >> 4) << 3;           // 0 or 8
    const _Float16* p = base + n * ld + kb;
    V16U r;
    r.h[0] = *(const v8h*)p;
    r.h[1] = *(const v8h*)(p + 16);
    return r.v;
}

// ---------------------------------------------------------------------------
// Elementwise fp32 -> f16 (q/k/v activations). One float4 per thread.
// ---------------------------------------------------------------------------
__global__ __launch_bounds__(256) void xcvt_kernel(const float* __restrict__ X,
                                                   _Float16* __restrict__ Xh)
{
    size_t gid = (size_t)blockIdx.x * 256 + threadIdx.x;
    float4 f = *(const float4*)(X + gid * 4);
    v4h o = {(_Float16)f.x, (_Float16)f.y, (_Float16)f.z, (_Float16)f.w};
    *(v4h*)(Xh + gid * 4) = o;
}

// ---------------------------------------------------------------------------
// Weight convert + transpose: W[K][N] fp32 -> Wt[N][K] f16 (64x64 LDS tiles).
// ---------------------------------------------------------------------------
__global__ __launch_bounds__(256) void wtrans_kernel(const float* __restrict__ W,
                                                     _Float16* __restrict__ Wt)
{
    __shared__ __align__(16) _Float16 Tl[64][72];
    const int tid = threadIdx.x;
    const int Bx = blockIdx.x;                 // N tile
    const int By = blockIdx.y;                 // K tile
    #pragma unroll
    for (int i = 0; i < 4; ++i) {
        int idx = tid + i * 256;               // 0..1023
        int r  = idx >> 4;                     // 0..63 (K)
        int c4 = (idx & 15) * 4;               // 0..60 (N)
        float4 f = *(const float4*)(W + (size_t)(By * 64 + r) * D_MODEL + Bx * 64 + c4);
        v4h t = {(_Float16)f.x, (_Float16)f.y, (_Float16)f.z, (_Float16)f.w};
        *(v4h*)&Tl[r][c4] = t;
    }
    __syncthreads();
    const int n  = tid >> 2;                   // 0..63
    const int kc = (tid & 3) * 16;             // 0,16,32,48
    v8h o0, o1;
    #pragma unroll
    for (int e = 0; e < 8; ++e) { o0[e] = Tl[kc + e][n]; o1[e] = Tl[kc + 8 + e][n]; }
    _Float16* dst = Wt + (size_t)(Bx * 64 + n) * D_MODEL + By * 64 + kc;
    *(v8h*)dst = o0;
    *(v8h*)(dst + 8) = o1;
}

// ---------------------------------------------------------------------------
// Projection GEMM (all f16 in LDS, fp32 accum): Y = (A * Wt^T + bias) * scale
// A[4096x1024] f16 row-major, Wt[1024(N)][1024(K)] f16.
// Epilogue scatters f16 into [bh][T][DK] (vtrans=0) or [bh][DK][T] (vtrans=1).
// Double-buffered LDS: one barrier per K-step.
// ---------------------------------------------------------------------------
__global__ __launch_bounds__(256) void gemm_proj(
    const _Float16* __restrict__ A, const _Float16* __restrict__ BtG,
    const float* __restrict__ bias, _Float16* __restrict__ Yh,
    float scale, int vtrans)
{
    __shared__ __align__(16) _Float16 As[2][128][40];
    __shared__ __align__(16) _Float16 Bs[2][128][40];

    const int tid  = threadIdx.x;
    const int lane = tid & 31;
    const int wave = tid >> 5;
    const int wm   = wave & 3;
    const int wn   = wave >> 2;
    const int Mt   = blockIdx.y * 128;
    const int Nt   = blockIdx.x * 128;

    auto stage = [&](int buf, int kt) {
        #pragma unroll
        for (int i = 0; i < 2; ++i) {
            int idx = tid + i * 256;           // 0..511
            int row = idx >> 2;                // 0..127
            int c8  = (idx & 3) * 8;           // 0,8,16,24
            *(v8h*)&As[buf][row][c8] =
                *(const v8h*)(A + (size_t)(Mt + row) * D_MODEL + kt + c8);
            *(v8h*)&Bs[buf][row][c8] =
                *(const v8h*)(BtG + (size_t)(Nt + row) * D_MODEL + kt + c8);
        }
        if (kt + 32 < D_MODEL) {
            __builtin_prefetch(A + (size_t)(Mt + (tid >> 1)) * D_MODEL + kt + 32, 0, 3);
            __builtin_prefetch(BtG + (size_t)(Nt + (tid >> 1)) * D_MODEL + kt + 32, 0, 3);
        }
    };

    v8f acc[2][4] = {};
    stage(0, 0);

    for (int kt = 0; kt < D_MODEL; kt += 32) {
        int buf = (kt >> 5) & 1;
        __syncthreads();
        if (kt + 32 < D_MODEL) stage(buf ^ 1, kt + 32);

        v16h af0 = frag_a(&As[buf][wm * 32 + 0][0], 40);
        v16h af1 = frag_a(&As[buf][wm * 32 + 16][0], 40);
        #pragma unroll
        for (int ns = 0; ns < 4; ++ns) {
            v16h bf = frag_b(&Bs[buf][wn * 64 + ns * 16][0], 40);
            acc[0][ns] = wmma16(af0, bf, acc[0][ns]);
            acc[1][ns] = wmma16(af1, bf, acc[1][ns]);
        }
    }

    #pragma unroll
    for (int ms = 0; ms < 2; ++ms) {
        #pragma unroll
        for (int ns = 0; ns < 4; ++ns) {
            int col = Nt + wn * 64 + ns * 16 + (lane & 15);
            float bv = bias[col];
            int h  = col >> 6;
            int dk = col & (DK - 1);
            #pragma unroll
            for (int r = 0; r < 8; ++r) {
                int row = Mt + wm * 32 + ms * 16 + r + ((lane >> 4) << 3);
                int bb  = row >> 11;
                int t   = row & (T_SEQ - 1);
                float y = (acc[ms][ns][r] + bv) * scale;
                size_t oidx = vtrans
                    ? ((size_t)(bb * NH + h) * DK + dk) * T_SEQ + t
                    : ((size_t)(bb * NH + h) * T_SEQ + t) * DK + dk;
                Yh[oidx] = (_Float16)y;
            }
        }
    }
}

// ---------------------------------------------------------------------------
// Flash attention: block = (b*h, 64 Q rows), 4 waves x 16 rows.
// K staged [key][dim] and V staged pre-transposed [dim][key] in LDS,
// double-buffered. Q pre-scaled by (1/sqrt(DK))*log2(e) -> exp2 softmax.
// ---------------------------------------------------------------------------
__global__ __launch_bounds__(128) void attn_kernel(
    const _Float16* __restrict__ Qh, const _Float16* __restrict__ Kh,
    const _Float16* __restrict__ VhT, _Float16* __restrict__ Out)
{
    __shared__ __align__(16) _Float16 Ks[2][32][72];   // [key][dim]
    __shared__ __align__(16) _Float16 Vt[2][64][40];   // [dim][key]
    __shared__ __align__(16) _Float16 Pl[4][16][40];   // per-wave P (16x32)

    const int tid  = threadIdx.x;
    const int lane = tid & 31;
    const int wave = tid >> 5;
    const int bh   = blockIdx.y;
    const int b    = bh >> 4;
    const int h    = bh & (NH - 1);
    const int qrow0 = blockIdx.x * 64 + wave * 16;

    const _Float16* Qb  = Qh + ((size_t)bh * T_SEQ + qrow0) * DK;
    const _Float16* Kb  = Kh + (size_t)bh * T_SEQ * DK;
    const _Float16* VbT = VhT + (size_t)bh * DK * T_SEQ;

    auto stageKV = [&](int buf, int j) {
        // K block: 32 keys x 64 dims, row-major copy (2 x b128 per thread)
        #pragma unroll
        for (int i = 0; i < 2; ++i) {
            int idx = tid + i * 128;           // 0..255
            int key = idx >> 3;                // 0..31
            int c8  = (idx & 7) * 8;           // 0..56
            *(v8h*)&Ks[buf][key][c8] =
                *(const v8h*)(Kb + (size_t)(j + key) * DK + c8);
        }
        // V^T block: 64 dims x 32 keys, contiguous copy (2 x b128 per thread)
        {
            int dim = tid >> 1;                // 0..63
            int kc  = (tid & 1) * 16;          // 0 or 16
            const _Float16* src = VbT + (size_t)dim * T_SEQ + j + kc;
            *(v8h*)&Vt[buf][dim][kc]     = *(const v8h*)src;
            *(v8h*)&Vt[buf][dim][kc + 8] = *(const v8h*)(src + 8);
        }
        if (j + 32 < T_SEQ) {
            __builtin_prefetch(Kb + (size_t)(j + 32 + (tid >> 2)) * DK, 0, 3);
            __builtin_prefetch(VbT + (size_t)(tid >> 1) * T_SEQ + j + 32, 0, 3);
        }
    };

    // Q fragments resident for whole kernel
    v16h qa0 = frag_a(Qb + 0, DK);
    v16h qa1 = frag_a(Qb + 32, DK);

    v8f O[4] = {};
    float mstat[8], lstat[8];
    #pragma unroll
    for (int r = 0; r < 8; ++r) { mstat[r] = -1e30f; lstat[r] = 0.0f; }

    stageKV(0, 0);

    for (int j = 0; j < T_SEQ; j += 32) {
        int buf = (j >> 5) & 1;
        __syncthreads();                       // Ks/Vt[buf] staged; Pl consumed
        if (j + 32 < T_SEQ) stageKV(buf ^ 1, j + 32);

        // S = Q * K^T (32 keys), K fragments from LDS
        v8f s0 = {}, s1 = {};
        s0 = wmma16(qa0, frag_b(&Ks[buf][0][0],   72), s0);
        s0 = wmma16(qa1, frag_b(&Ks[buf][0][32],  72), s0);
        s1 = wmma16(qa0, frag_b(&Ks[buf][16][0],  72), s1);
        s1 = wmma16(qa1, frag_b(&Ks[buf][16][32], 72), s1);

        // Online softmax (rows rbase..rbase+7 per lane group)
        float p0[8], p1[8], alpha[8];
        #pragma unroll
        for (int r = 0; r < 8; ++r) {
            float mx = fmaxf(s0[r], s1[r]);
            #pragma unroll
            for (int msk = 8; msk >= 1; msk >>= 1)
                mx = fmaxf(mx, __shfl_xor(mx, msk, 32));
            float mnew = fmaxf(mstat[r], mx);
            alpha[r] = __builtin_exp2f(mstat[r] - mnew);
            p0[r]    = __builtin_exp2f(s0[r] - mnew);
            p1[r]    = __builtin_exp2f(s1[r] - mnew);
            float rs = p0[r] + p1[r];
            #pragma unroll
            for (int msk = 8; msk >= 1; msk >>= 1)
                rs += __shfl_xor(rs, msk, 32);
            lstat[r] = lstat[r] * alpha[r] + rs;
            mstat[r] = mnew;
        }
        #pragma unroll
        for (int t = 0; t < 4; ++t)
            #pragma unroll
            for (int r = 0; r < 8; ++r)
                O[t][r] *= alpha[r];

        // C-layout P -> LDS (A-fragment source layout)
        {
            int col   = lane & 15;
            int rbase = (lane >> 4) << 3;
            #pragma unroll
            for (int r = 0; r < 8; ++r) {
                Pl[wave][rbase + r][col]      = (_Float16)p0[r];
                Pl[wave][rbase + r][16 + col] = (_Float16)p1[r];
            }
        }
        __syncthreads();                       // Pl visible

        // O += P * V
        v16h pf = frag_a(&Pl[wave][0][0], 40);
        #pragma unroll
        for (int t = 0; t < 4; ++t) {
            v16h vf = frag_b(&Vt[buf][t * 16][0], 40);
            O[t] = wmma16(pf, vf, O[t]);
        }
    }

    // Normalize, write f16 attention output [B][T][D_MODEL]
    {
        int col   = lane & 15;
        int rbase = (lane >> 4) << 3;
        #pragma unroll
        for (int r = 0; r < 8; ++r) {
            float inv = 1.0f / lstat[r];
            int trow = qrow0 + rbase + r;
            #pragma unroll
            for (int t = 0; t < 4; ++t) {
                Out[((size_t)b * T_SEQ + trow) * D_MODEL + h * DK + t * 16 + col] =
                    (_Float16)(O[t][r] * inv);
            }
        }
    }
}

// ---------------------------------------------------------------------------
// Output projection: Y fp32 = Xh f16 [4096x1024] * Wt_o^T + b_o
// ---------------------------------------------------------------------------
__global__ __launch_bounds__(256) void gemm_out(
    const _Float16* __restrict__ A, const _Float16* __restrict__ BtG,
    const float* __restrict__ bias, float* __restrict__ Y)
{
    __shared__ __align__(16) _Float16 As[2][128][40];
    __shared__ __align__(16) _Float16 Bs[2][128][40];

    const int tid  = threadIdx.x;
    const int lane = tid & 31;
    const int wave = tid >> 5;
    const int wm   = wave & 3;
    const int wn   = wave >> 2;
    const int Mt   = blockIdx.y * 128;
    const int Nt   = blockIdx.x * 128;

    auto stage = [&](int buf, int kt) {
        #pragma unroll
        for (int i = 0; i < 2; ++i) {
            int idx = tid + i * 256;
            int row = idx >> 2;
            int c8  = (idx & 3) * 8;
            *(v8h*)&As[buf][row][c8] =
                *(const v8h*)(A + (size_t)(Mt + row) * D_MODEL + kt + c8);
            *(v8h*)&Bs[buf][row][c8] =
                *(const v8h*)(BtG + (size_t)(Nt + row) * D_MODEL + kt + c8);
        }
        if (kt + 32 < D_MODEL) {
            __builtin_prefetch(A + (size_t)(Mt + (tid >> 1)) * D_MODEL + kt + 32, 0, 3);
            __builtin_prefetch(BtG + (size_t)(Nt + (tid >> 1)) * D_MODEL + kt + 32, 0, 3);
        }
    };

    v8f acc[2][4] = {};
    stage(0, 0);

    for (int kt = 0; kt < D_MODEL; kt += 32) {
        int buf = (kt >> 5) & 1;
        __syncthreads();
        if (kt + 32 < D_MODEL) stage(buf ^ 1, kt + 32);

        v16h af0 = frag_a(&As[buf][wm * 32 + 0][0], 40);
        v16h af1 = frag_a(&As[buf][wm * 32 + 16][0], 40);
        #pragma unroll
        for (int ns = 0; ns < 4; ++ns) {
            v16h bf = frag_b(&Bs[buf][wn * 64 + ns * 16][0], 40);
            acc[0][ns] = wmma16(af0, bf, acc[0][ns]);
            acc[1][ns] = wmma16(af1, bf, acc[1][ns]);
        }
    }

    #pragma unroll
    for (int ms = 0; ms < 2; ++ms) {
        #pragma unroll
        for (int ns = 0; ns < 4; ++ns) {
            int col = Nt + wn * 64 + ns * 16 + (lane & 15);
            float bv = bias[col];
            #pragma unroll
            for (int r = 0; r < 8; ++r) {
                int row = Mt + wm * 32 + ms * 16 + r + ((lane >> 4) << 3);
                Y[(size_t)row * D_MODEL + col] = acc[ms][ns][r] + bv;
            }
        }
    }
}

// ---------------------------------------------------------------------------
extern "C" void kernel_launch(void* const* d_in, const int* in_sizes, int n_in,
                              void* d_out, int out_size, void* d_ws, size_t ws_size,
                              hipStream_t stream) {
    const float* q   = (const float*)d_in[0];
    const float* k   = (const float*)d_in[1];
    const float* v   = (const float*)d_in[2];
    const float* w_q = (const float*)d_in[3];
    const float* b_q = (const float*)d_in[4];
    const float* w_k = (const float*)d_in[5];
    const float* b_k = (const float*)d_in[6];
    const float* w_v = (const float*)d_in[7];
    const float* b_v = (const float*)d_in[8];
    const float* w_o = (const float*)d_in[9];
    const float* b_o = (const float*)d_in[10];
    float* out = (float*)d_out;

    const size_t XE = (size_t)2 * T_SEQ * D_MODEL;        // 4 Mi activations
    const size_t WE = (size_t)D_MODEL * D_MODEL;          // 1 Mi weights
    _Float16* Xq  = (_Float16*)d_ws;
    _Float16* Xk  = Xq + XE;
    _Float16* Xv  = Xk + XE;
    _Float16* Wtq = Xv + XE;
    _Float16* Wtk = Wtq + WE;
    _Float16* Wtv = Wtk + WE;
    _Float16* Wto = Wtv + WE;
    _Float16* qh  = Wto + WE;          // [bh][T][DK]
    _Float16* kh  = qh + XE;           // [bh][T][DK]
    _Float16* vhT = kh + XE;           // [bh][DK][T]
    _Float16* ao  = vhT + XE;          // [B*T][D_MODEL]

    // Activation + weight precision/layout prep (one-shot, bandwidth-trivial)
    const int xblocks = (int)(XE / 4 / 256);              // 4096
    xcvt_kernel<<<xblocks, 256, 0, stream>>>(q, Xq);
    xcvt_kernel<<<xblocks, 256, 0, stream>>>(k, Xk);
    xcvt_kernel<<<xblocks, 256, 0, stream>>>(v, Xv);
    dim3 gW(D_MODEL / 64, D_MODEL / 64);                  // (16,16)
    wtrans_kernel<<<gW, 256, 0, stream>>>(w_q, Wtq);
    wtrans_kernel<<<gW, 256, 0, stream>>>(w_k, Wtk);
    wtrans_kernel<<<gW, 256, 0, stream>>>(w_v, Wtv);
    wtrans_kernel<<<gW, 256, 0, stream>>>(w_o, Wto);

    // Fold softmax scale and log2(e) into Q so softmax uses exp2 directly.
    const float qscale = 0.125f * 1.44269504088896340736f;

    dim3 gP(D_MODEL / 128, (2 * T_SEQ) / 128);            // (8, 32)
    gemm_proj<<<gP, dim3(256), 0, stream>>>(Xq, Wtq, b_q, qh, qscale, 0);
    gemm_proj<<<gP, dim3(256), 0, stream>>>(Xk, Wtk, b_k, kh, 1.0f, 0);
    gemm_proj<<<gP, dim3(256), 0, stream>>>(Xv, Wtv, b_v, vhT, 1.0f, 1);

    dim3 gA(T_SEQ / 64, 2 * NH);                          // (32, 32)
    attn_kernel<<<gA, dim3(128), 0, stream>>>(qh, kh, vhT, ao);

    gemm_out<<<gP, dim3(256), 0, stream>>>(ao, Wto, b_o, out);
}
